// FullEncoder_43542378447039
// MI455X (gfx1250) — matrix operand order repaired and checked by reading, not compiled, not run on hardware
//
#include <hip/hip_runtime.h>

typedef _Float16 half_t;
typedef __attribute__((ext_vector_type(16))) _Float16 v16h;
typedef __attribute__((ext_vector_type(8)))  _Float16 v8h;
typedef __attribute__((ext_vector_type(8)))  float    v8f;

#define DEV_INLINE __device__ __forceinline__

constexpr int B_   = 4;
constexpr int S_   = 2048;
constexpr int D_   = 512;
constexpr int HID_ = 512;
constexpr int H_   = 8;
constexpr int DFF_ = 2048;
constexpr int DK_  = 64;
constexpr int DV_  = 64;
constexpr int NTOK = B_ * S_;   // 8192 rows for all token GEMMs

// ---------------------------------------------------------------------------
// WMMA fragment loaders (CDNA5 wave32 layouts, CDNA5 ISA §7.12.2)
//
// A (16x32, f16, row-major src, leading dim ld):
//   lanes 0-15 : row M=lane, slots 0..7 -> K=0..7,  slots 8..15 -> K=16..23
//   lanes16-31 : row M=lane-16, slots -> K=8..15 and K=24..31
// => two contiguous 16B loads per lane (base, base+32B).
DEV_INLINE v16h load_a_frag(const half_t* p, int ld) {
  const int lane = threadIdx.x & 31;
  const int hs   = lane >> 4;
  const int r    = lane & 15;
  const half_t* base = p + (size_t)r * ld + hs * 8;
  v8h lo = *(const v8h*)(base);
  v8h hi = *(const v8h*)(base + 16);
  v16h f;
#pragma unroll
  for (int i = 0; i < 8; ++i) { f[i] = lo[i]; f[8 + i] = hi[i]; }
  return f;
}

// B (32x16, f16) loaded from B^T stored [N,K] row-major (leading dim ld):
//   lane (hs,n): slots 0..7 -> K=hs*8..+7, slots 8..15 -> K=16+hs*8..+7 (same as A)
// => two contiguous 16B loads per lane (base, base+32B).
DEV_INLINE v16h load_bt_frag(const half_t* p, int ld) {
  const int lane = threadIdx.x & 31;
  const int hs   = lane >> 4;
  const int n    = lane & 15;
  const half_t* base = p + (size_t)n * ld + hs * 8;
  v8h lo = *(const v8h*)(base);
  v8h hi = *(const v8h*)(base + 16);
  v16h f;
#pragma unroll
  for (int i = 0; i < 8; ++i) { f[i] = lo[i]; f[8 + i] = hi[i]; }
  return f;
}

DEV_INLINE v8f wmma_f16(v16h a, v16h b, v8f c) {
  return __builtin_amdgcn_wmma_f32_16x16x32_f16(false, a, false, b, (short)0, c,
                                                false, false);
}

// ---------------------------------------------------------------------------
// 1) Embedding gather + sinusoidal positional encoding
__global__ __launch_bounds__(256) void embed_pe_kernel(
    const int* __restrict__ x, const float* __restrict__ emb,
    float* __restrict__ z32, half_t* __restrict__ z16) {
  const int tok = blockIdx.x;            // 0..NTOK-1
  const int s   = tok % S_;
  const int id  = x[tok];
  const float kLog1e4 = 9.210340371976184f;  // ln(10000)
  for (int d = threadIdx.x; d < D_; d += blockDim.x) {
    float ang = (float)s * __expf(-((float)d / (float)D_) * kLog1e4);
    float pe  = (d & 1) ? __cosf(ang) : __sinf(ang);
    float v   = emb[(size_t)id * D_ + d] + pe;
    z32[(size_t)tok * D_ + d] = v;
    z16[(size_t)tok * D_ + d] = (half_t)v;
  }
}

// 2) Weight converters: f32 [K,N] -> f16 [N,K] (transposed for B-fragment loads)
__global__ __launch_bounds__(256) void w_transpose_f16(
    const float* __restrict__ src, half_t* __restrict__ dst, int K, int N) {
  int idx = blockIdx.x * 256 + threadIdx.x;
  if (idx >= K * N) return;
  int n = idx / K, k = idx % K;
  dst[idx] = (half_t)src[(size_t)k * N + n];
}

// Headed weights (H, Din, Dk) -> f16 [N=H*Dk, Din]
__global__ __launch_bounds__(256) void w_transpose_headed_f16(
    const float* __restrict__ src, half_t* __restrict__ dst,
    int nh, int Din, int Dk) {
  int idx = blockIdx.x * 256 + threadIdx.x;
  if (idx >= nh * Din * Dk) return;
  int n = idx / Din, d = idx % Din;
  int h = n / Dk, k = n % Dk;
  dst[idx] = (half_t)src[((size_t)h * Din + d) * Dk + k];
}

// 3) Generic WMMA GEMM:  C[M,N] = A_f16[M,K] * Bt_f16[N,K]^T + bias (+relu)
//    256 threads = 8 waves; block tile 128x128; wave tile 32x64 (2x4 frags,
//    8 v_wmma per K-step => ~21 FLOP per fetched byte).
DEV_INLINE void gemm_store_tile(v8f c, int m0, int n0, int N,
                                const float* bias, float* o32, half_t* o16,
                                int relu) {
  const int lane = threadIdx.x & 31;
  const int hs = lane >> 4, nn = lane & 15;
  const int col = n0 + nn;
  const float bv = bias ? bias[col] : 0.0f;
#pragma unroll
  for (int r = 0; r < 8; ++r) {
    int row = m0 + hs * 8 + r;           // C layout: lanes0-15 M=r, lanes16-31 M=8+r
    float v = c[r] + bv;
    if (relu) v = fmaxf(v, 0.0f);
    if (o32) o32[(size_t)row * N + col] = v;
    if (o16) o16[(size_t)row * N + col] = (half_t)v;
  }
}

__global__ __launch_bounds__(256) void gemm_wmma_kernel(
    const half_t* __restrict__ A, const half_t* __restrict__ Bt,
    const float* __restrict__ bias, float* __restrict__ out32,
    half_t* __restrict__ out16, int M, int N, int Kd, int relu) {
  const int lane = threadIdx.x & 31;
  const int wave = threadIdx.x >> 5;
  const int wm = wave & 3;               // 4 waves along M (4 * 32 = 128)
  const int wn = wave >> 2;              // 2 waves along N (2 * 64 = 128)
  const int m0 = blockIdx.x * 128 + wm * 32;
  const int n0 = blockIdx.y * 128 + wn * 64;

  const half_t* a0p = A + (size_t)m0 * Kd;
  const half_t* a1p = A + (size_t)(m0 + 16) * Kd;
  const half_t* b0p = Bt + (size_t)(n0 +  0) * Kd;
  const half_t* b1p = Bt + (size_t)(n0 + 16) * Kd;
  const half_t* b2p = Bt + (size_t)(n0 + 32) * Kd;
  const half_t* b3p = Bt + (size_t)(n0 + 48) * Kd;

  v8f acc[2][4];
#pragma unroll
  for (int i = 0; i < 2; ++i)
#pragma unroll
    for (int j = 0; j < 4; ++j) acc[i][j] = v8f{0, 0, 0, 0, 0, 0, 0, 0};

  const size_t prow = (size_t)(lane & 15) * Kd;   // per-lane prefetch row
  for (int k0 = 0; k0 < Kd; k0 += 32) {
    // prefetch the streaming A operand eight k-tiles ahead; locality=3 =>
    // WGP scope (pull into all cache levels); speculative, OOB dropped.
    __builtin_prefetch(a0p + prow + k0 + 256, 0, 3);
    __builtin_prefetch(a1p + prow + k0 + 256, 0, 3);

    v16h a0 = load_a_frag(a0p + k0, Kd);
    v16h a1 = load_a_frag(a1p + k0, Kd);
    v16h b0 = load_bt_frag(b0p + k0, Kd);
    v16h b1 = load_bt_frag(b1p + k0, Kd);
    v16h b2 = load_bt_frag(b2p + k0, Kd);
    v16h b3 = load_bt_frag(b3p + k0, Kd);
    acc[0][0] = wmma_f16(a0, b0, acc[0][0]);
    acc[0][1] = wmma_f16(a0, b1, acc[0][1]);
    acc[0][2] = wmma_f16(a0, b2, acc[0][2]);
    acc[0][3] = wmma_f16(a0, b3, acc[0][3]);
    acc[1][0] = wmma_f16(a1, b0, acc[1][0]);
    acc[1][1] = wmma_f16(a1, b1, acc[1][1]);
    acc[1][2] = wmma_f16(a1, b2, acc[1][2]);
    acc[1][3] = wmma_f16(a1, b3, acc[1][3]);
  }
#pragma unroll
  for (int i = 0; i < 2; ++i)
#pragma unroll
    for (int j = 0; j < 4; ++j)
      gemm_store_tile(acc[i][j], m0 + i * 16, n0 + j * 16, N, bias, out32,
                      out16, relu);
}

// 4) V transpose for PV fragments: [b,s,h,v] -> [b,h,v,s]
__global__ __launch_bounds__(256) void v_transpose_kernel(
    const half_t* __restrict__ vh, half_t* __restrict__ vt) {
  int idx = blockIdx.x * 256 + threadIdx.x;
  if (idx >= B_ * S_ * D_) return;
  int v = idx % DV_;
  int h = (idx / DV_) % H_;
  int s = (idx / (DV_ * H_)) % S_;
  int b = idx / (DV_ * H_ * S_);
  vt[(((size_t)(b * H_ + h) * DV_) + v) * S_ + s] = vh[idx];
}

// 5) Flash attention: one wave per 16-query tile; online softmax; 8 WMMAs
//    per 32-key block (4 for QK^T over DK=64, 4 for P@V over DV=64).
__global__ __launch_bounds__(128) void flash_attn_kernel(
    const half_t* __restrict__ Qh, const half_t* __restrict__ Kh,
    const half_t* __restrict__ Vt, half_t* __restrict__ outc) {
  const int lane = threadIdx.x & 31;
  const int wave = threadIdx.x >> 5;
  const int bhq = blockIdx.x * 4 + wave;           // B*H*(S/16) = 4096 tiles
  const int qt = bhq & 127;                        // S/16 = 128
  const int h  = (bhq >> 7) & 7;
  const int b  = bhq >> 10;
  const int q0 = qt * 16;
  const int ldq = H_ * DK_;                        // 512 (layout [b,s,h,k])

  const half_t* qbase = Qh + ((size_t)(b * S_ + q0) * H_ + h) * DK_;
  const half_t* kbase = Kh + ((size_t)b * S_ * H_ + h) * DK_;   // + s*ldq
  const half_t* vbase = Vt + ((size_t)(b * H_ + h) * DV_) * S_; // + n*S_ + s

  v16h qf0 = load_a_frag(qbase, ldq);          // K = 0..31
  v16h qf1 = load_a_frag(qbase + 32, ldq);     // K = 32..63

  v8f o0 = {0,0,0,0,0,0,0,0}, o1 = {0,0,0,0,0,0,0,0};
  v8f o2 = {0,0,0,0,0,0,0,0}, o3 = {0,0,0,0,0,0,0,0};
  float mrow[8], lrow[8];
#pragma unroll
  for (int r = 0; r < 8; ++r) { mrow[r] = -1e30f; lrow[r] = 0.0f; }

  __shared__ __align__(32) half_t plds[4][16 * 32];
  half_t* pl = &plds[wave][0];
  const int hs = lane >> 4, nn = lane & 15;
  const float scale = 0.125f;   // 1/sqrt(DK)

  for (int kb = 0; kb < S_ / 32; ++kb) {
    const int s0 = kb * 32;
    // prefetch the next key/value block into WGP-adjacent cache
    __builtin_prefetch(kbase + (size_t)(s0 + 32 + nn) * ldq, 0, 3);
    __builtin_prefetch(vbase + (size_t)nn * S_ + s0 + 32, 0, 3);

    // ---- scores = Q * K^T for keys [s0, s0+32)
    v8f sc0 = {0,0,0,0,0,0,0,0}, sc1 = {0,0,0,0,0,0,0,0};
    sc0 = wmma_f16(qf0, load_bt_frag(kbase + (size_t)s0 * ldq, ldq), sc0);
    sc0 = wmma_f16(qf1, load_bt_frag(kbase + (size_t)s0 * ldq + 32, ldq), sc0);
    sc1 = wmma_f16(qf0, load_bt_frag(kbase + (size_t)(s0 + 16) * ldq, ldq), sc1);
    sc1 = wmma_f16(qf1, load_bt_frag(kbase + (size_t)(s0 + 16) * ldq + 32, ldq), sc1);

    // ---- online softmax (row reductions across the 16-lane half)
#pragma unroll
    for (int r = 0; r < 8; ++r) {
      float x0 = sc0[r] * scale, x1 = sc1[r] * scale;
      float mx = fmaxf(x0, x1);
      mx = fmaxf(mx, __shfl_xor(mx, 1, 32));
      mx = fmaxf(mx, __shfl_xor(mx, 2, 32));
      mx = fmaxf(mx, __shfl_xor(mx, 4, 32));
      mx = fmaxf(mx, __shfl_xor(mx, 8, 32));
      float mnew  = fmaxf(mrow[r], mx);
      float alpha = __expf(mrow[r] - mnew);
      float p0 = __expf(x0 - mnew);
      float p1 = __expf(x1 - mnew);
      float rs = p0 + p1;
      rs += __shfl_xor(rs, 1, 32);
      rs += __shfl_xor(rs, 2, 32);
      rs += __shfl_xor(rs, 4, 32);
      rs += __shfl_xor(rs, 8, 32);
      lrow[r] = lrow[r] * alpha + rs;
      mrow[r] = mnew;
      o0[r] *= alpha; o1[r] *= alpha; o2[r] *= alpha; o3[r] *= alpha;
      const int row = hs * 8 + r;
      pl[row * 32 + nn]      = (half_t)p0;   // keys s0..s0+15
      pl[row * 32 + nn + 16] = (half_t)p1;   // keys s0+16..s0+31
    }
    asm volatile("" ::: "memory");   // keep LDS P-store before P-load (same-wave LDS is in-order)

    // ---- O += P * V  (A = P 16x32 from LDS, B^T rows = Vt[n][s])
    v16h pa = load_a_frag(pl, 32);
    o0 = wmma_f16(pa, load_bt_frag(vbase + (size_t)0  * S_ + s0, S_), o0);
    o1 = wmma_f16(pa, load_bt_frag(vbase + (size_t)16 * S_ + s0, S_), o1);
    o2 = wmma_f16(pa, load_bt_frag(vbase + (size_t)32 * S_ + s0, S_), o2);
    o3 = wmma_f16(pa, load_bt_frag(vbase + (size_t)48 * S_ + s0, S_), o3);
  }

  // ---- normalize + write concat [b,s,h*DV+v]
#pragma unroll
  for (int r = 0; r < 8; ++r) {
    const int q = q0 + hs * 8 + r;
    const float inv = 1.0f / lrow[r];
    const size_t base = ((size_t)(b * S_ + q) * H_ + h) * DV_;
    outc[base + nn]      = (half_t)(o0[r] * inv);
    outc[base + 16 + nn] = (half_t)(o1[r] * inv);
    outc[base + 32 + nn] = (half_t)(o2[r] * inv);
    outc[base + 48 + nn] = (half_t)(o3[r] * inv);
  }
}

// 6) Fused residual-add + LayerNorm (+optional relu); one block per token row
__global__ __launch_bounds__(128) void add_ln_kernel(
    const float* __restrict__ xin, const float* __restrict__ res,
    const float* __restrict__ gamma, const float* __restrict__ beta,
    float* __restrict__ out32, half_t* __restrict__ out16, int relu) {
  const int row = blockIdx.x;
  const int tid = threadIdx.x;
  const float* px = xin + (size_t)row * D_;
  const float* pr = res + (size_t)row * D_;
  float v[4], s = 0.0f, sq = 0.0f;
#pragma unroll
  for (int j = 0; j < 4; ++j) {
    int d = tid + j * 128;
    float t = px[d] + pr[d];
    v[j] = t; s += t; sq += t * t;
  }
  __shared__ float ssum[128], ssq[128];
  ssum[tid] = s; ssq[tid] = sq;
  __syncthreads();
  for (int st = 64; st > 0; st >>= 1) {
    if (tid < st) { ssum[tid] += ssum[tid + st]; ssq[tid] += ssq[tid + st]; }
    __syncthreads();
  }
  const float mu   = ssum[0] * (1.0f / D_);
  const float var  = ssq[0] * (1.0f / D_) - mu * mu;
  const float rstd = rsqrtf(var + 1e-5f);
#pragma unroll
  for (int j = 0; j < 4; ++j) {
    int d = tid + j * 128;
    float t = (v[j] - mu) * rstd * gamma[d] + beta[d];
    if (relu) t = fmaxf(t, 0.0f);
    if (out32) out32[(size_t)row * D_ + d] = t;
    if (out16) out16[(size_t)row * D_ + d] = (half_t)t;
  }
}

// ---------------------------------------------------------------------------
extern "C" void kernel_launch(void* const* d_in, const int* in_sizes, int n_in,
                              void* d_out, int out_size, void* d_ws, size_t ws_size,
                              hipStream_t stream) {
  const int*   x    = (const int*)  d_in[0];
  const float* emb  = (const float*)d_in[1];
  const float* WfQ  = (const float*)d_in[2];  const float* bfQ = (const float*)d_in[3];
  const float* WfK  = (const float*)d_in[4];  const float* bfK = (const float*)d_in[5];
  const float* WfV  = (const float*)d_in[6];  const float* bfV = (const float*)d_in[7];
  const float* WQ   = (const float*)d_in[8];  const float* bQ  = (const float*)d_in[9];
  const float* WK   = (const float*)d_in[10]; const float* bK  = (const float*)d_in[11];
  const float* WV   = (const float*)d_in[12]; const float* bV  = (const float*)d_in[13];
  const float* Wo   = (const float*)d_in[14]; const float* bo  = (const float*)d_in[15];
  const float* W1   = (const float*)d_in[16]; const float* b1  = (const float*)d_in[17];
  const float* W2   = (const float*)d_in[18]; const float* b2  = (const float*)d_in[19];
  const float* gamma = (const float*)d_in[20];
  const float* beta  = (const float*)d_in[21];
  float* out = (float*)d_out;

  // --- carve workspace ---
  size_t off = 0;
  auto carve = [&](size_t bytes) -> void* {
    void* p = (char*)d_ws + off;
    off += (bytes + 255) & ~(size_t)255;
    return p;
  };
  float*  z32   = (float*) carve((size_t)NTOK * D_ * 4);
  half_t* z16   = (half_t*)carve((size_t)NTOK * D_ * 2);
  half_t* wfqt  = (half_t*)carve((size_t)D_ * HID_ * 2);
  half_t* wfkt  = (half_t*)carve((size_t)D_ * HID_ * 2);
  half_t* wfvt  = (half_t*)carve((size_t)D_ * D_ * 2);
  half_t* wqt   = (half_t*)carve((size_t)HID_ * HID_ * 2);
  half_t* wkt   = (half_t*)carve((size_t)HID_ * HID_ * 2);
  half_t* wvt   = (half_t*)carve((size_t)D_ * D_ * 2);
  half_t* wot   = (half_t*)carve((size_t)D_ * D_ * 2);
  half_t* w1t   = (half_t*)carve((size_t)D_ * DFF_ * 2);
  half_t* w2t   = (half_t*)carve((size_t)DFF_ * D_ * 2);
  half_t* q16   = (half_t*)carve((size_t)NTOK * HID_ * 2);
  half_t* k16   = (half_t*)carve((size_t)NTOK * HID_ * 2);
  half_t* v16   = (half_t*)carve((size_t)NTOK * D_ * 2);
  half_t* qh    = (half_t*)carve((size_t)NTOK * HID_ * 2);
  half_t* kh    = (half_t*)carve((size_t)NTOK * HID_ * 2);
  half_t* vh    = (half_t*)carve((size_t)NTOK * D_ * 2);
  float*  attn32= (float*) carve((size_t)NTOK * D_ * 4);
  float*  a32   = (float*) carve((size_t)NTOK * D_ * 4);
  half_t* f1    = (half_t*)carve((size_t)NTOK * DFF_ * 2);
  // aliases of buffers that are dead by the time these are written:
  half_t* vt     = q16;     // q16 consumed by qh-GEMM before vt is produced
  half_t* concat = k16;     // k16 consumed by kh-GEMM before flash writes concat
  half_t* a16    = v16;     // v16 consumed by vh-GEMM before LN writes a16
  float*  f2     = attn32;  // attn32 consumed by first LN before f2-GEMM

  const int T256 = 256;
  auto g1 = [&](size_t n) { return dim3((unsigned)((n + T256 - 1) / T256)); };

  // 1) embeddings + positional encoding
  embed_pe_kernel<<<NTOK, 256, 0, stream>>>(x, emb, z32, z16);

  // 2) weight conversion (f32 -> transposed f16)
  w_transpose_f16<<<g1((size_t)D_ * HID_), 256, 0, stream>>>(WfQ, wfqt, D_, HID_);
  w_transpose_f16<<<g1((size_t)D_ * HID_), 256, 0, stream>>>(WfK, wfkt, D_, HID_);
  w_transpose_f16<<<g1((size_t)D_ * D_),   256, 0, stream>>>(WfV, wfvt, D_, D_);
  w_transpose_headed_f16<<<g1((size_t)H_ * HID_ * DK_), 256, 0, stream>>>(WQ, wqt, H_, HID_, DK_);
  w_transpose_headed_f16<<<g1((size_t)H_ * HID_ * DK_), 256, 0, stream>>>(WK, wkt, H_, HID_, DK_);
  w_transpose_headed_f16<<<g1((size_t)H_ * D_ * DV_),   256, 0, stream>>>(WV, wvt, H_, D_, DV_);
  w_transpose_f16<<<g1((size_t)D_ * D_),    256, 0, stream>>>(Wo, wot, D_, D_);
  w_transpose_f16<<<g1((size_t)D_ * DFF_),  256, 0, stream>>>(W1, w1t, D_, DFF_);
  w_transpose_f16<<<g1((size_t)DFF_ * D_),  256, 0, stream>>>(W2, w2t, DFF_, D_);

  // 3) fused-dim projections Q,K,V  (8192 x 512 x 512)
  dim3 blk(256);
  dim3 grd512(NTOK / 128, HID_ / 128);        // 64 x 4 blocks
  gemm_wmma_kernel<<<grd512, blk, 0, stream>>>(z16, wfqt, bfQ, nullptr, q16, NTOK, HID_, D_, 0);
  gemm_wmma_kernel<<<grd512, blk, 0, stream>>>(z16, wfkt, bfK, nullptr, k16, NTOK, HID_, D_, 0);
  gemm_wmma_kernel<<<grd512, blk, 0, stream>>>(z16, wfvt, bfV, nullptr, v16, NTOK, D_,   D_, 0);

  // 4) per-head projections (heads concatenated along N => [b,s,h,dk] layout)
  gemm_wmma_kernel<<<grd512, blk, 0, stream>>>(q16, wqt, bQ, nullptr, qh, NTOK, HID_, HID_, 0);
  gemm_wmma_kernel<<<grd512, blk, 0, stream>>>(k16, wkt, bK, nullptr, kh, NTOK, HID_, HID_, 0);
  gemm_wmma_kernel<<<grd512, blk, 0, stream>>>(v16, wvt, bV, nullptr, vh, NTOK, D_,   D_,   0);

  // 5) V transpose for PV fragment loads, then flash attention
  v_transpose_kernel<<<g1((size_t)B_ * S_ * D_), 256, 0, stream>>>(vh, vt);
  flash_attn_kernel<<<(B_ * H_ * (S_ / 16)) / 4, 128, 0, stream>>>(qh, kh, vt, concat);

  // 6) output projection + first residual LayerNorm
  gemm_wmma_kernel<<<grd512, blk, 0, stream>>>(concat, wot, bo, attn32, nullptr, NTOK, D_, D_, 0);
  add_ln_kernel<<<NTOK, 128, 0, stream>>>(attn32, z32, gamma, beta, a32, a16, 0);

  // 7) FFN: relu(a@W1+b1)@W2+b2, second LayerNorm, final relu -> d_out
  dim3 grd_ffn1(NTOK / 128, DFF_ / 128);      // 64 x 16 blocks
  gemm_wmma_kernel<<<grd_ffn1, blk, 0, stream>>>(a16, w1t, b1, nullptr, f1, NTOK, DFF_, D_, 1);
  gemm_wmma_kernel<<<grd512, blk, 0, stream>>>(f1, w2t, b2, f2, nullptr, NTOK, D_, DFF_, 0);
  add_ln_kernel<<<NTOK, 128, 0, stream>>>(f2, a32, gamma, beta, out, nullptr, 1);

  (void)in_sizes; (void)n_in; (void)out_size; (void)ws_size;
}
